// Fetal_Pres_Net_82274393522995
// MI455X (gfx1250) — matrix operand order repaired and checked by reading, not compiled, not run on hardware
//
#include <hip/hip_runtime.h>
#include <hip/hip_bf16.h>

// ---------------- types ----------------
typedef __attribute__((ext_vector_type(16))) __bf16 v16bf;
typedef __attribute__((ext_vector_type(8)))  float  v8f;

union BFU {
  v16bf v;
  unsigned u32[8];
  unsigned short u[16];
};

// pack two f32 into one dword of two bf16 (truncate) with a single v_perm_b32:
// low 16 = bf16(lo), high 16 = bf16(hi)
__device__ __forceinline__ unsigned pkbf(float lo, float hi) {
  return __builtin_amdgcn_perm(__float_as_uint(hi), __float_as_uint(lo), 0x07060302u);
}

__device__ __forceinline__ v8f vzero8() {
  v8f z;
#pragma unroll
  for (int i = 0; i < 8; ++i) z[i] = 0.0f;
  return z;
}

// reduce across the 16 lanes of a wave half (xor masks 1..8 stay inside halves)
__device__ __forceinline__ float redmax16(float x) {
#pragma unroll
  for (int m = 1; m < 16; m <<= 1) x = fmaxf(x, __shfl_xor(x, m, 32));
  return x;
}
__device__ __forceinline__ float redsum16(float x) {
#pragma unroll
  for (int m = 1; m < 16; m <<= 1) x += __shfl_xor(x, m, 32);
  return x;
}

// B=16, S=16, T=128, D=128, ST=2048
#define NB 16
#define NS 16
#define NT 128
#define ND 128

__global__ void fetal_zero_kernel(float* sums) {
  if (threadIdx.x < NB * 3) sums[threadIdx.x] = 0.0f;
}

__global__ __launch_bounds__(256, 1)
void fetal_attn_kernel(const float* __restrict__ q,
                       const float* __restrict__ k,
                       const float* __restrict__ v,
                       const float* __restrict__ Wc,
                       const float* __restrict__ bc,
                       const unsigned char* __restrict__ maskp,
                       const int* __restrict__ c1,
                       float* __restrict__ sums) {
  const int tid  = threadIdx.x;
  const int w    = tid >> 5;        // wave id 0..7 -> rows w*16..w*16+15
  const int lane = tid & 31;
  const int n    = lane & 15;
  const int hf   = lane >> 4;       // lane half (WMMA layout selector)
  const int b    = blockIdx.x >> 4;
  const int s    = blockIdx.x & 15;

  // K block (then V block) pre-converted to bf16, pre-packed as B-operand dwords:
  // layout [outer][64] where outer = t' (K) or d (V), dword j = {bf16(pair_hi), bf16(pair_lo)}
  __shared__ __align__(16) unsigned sKV[NT * (ND / 2)];   // 32KB
  __shared__ __align__(16) float sP[8][16 * NT];          // 64KB: per-wave P scratch (f32)

  // Fold 1/sqrt(D) AND log2(e) into Q: scores come out in base-2 domain,
  // so softmax uses a single v_exp_f32 (exp2) per element.
  const float qscale = 0.08838834764831845f * 1.4426950408889634f;

  // ---- Q rows (16 per wave) as bf16 A-operands, 4 K-chunks of 32 ----
  BFU qa[4];
  {
    const float* qrow = q + (((size_t)(b * NS + s)) * NT + (w * 16 + n)) * ND;
#pragma unroll
    for (int kc = 0; kc < 4; ++kc)
#pragma unroll
      for (int j = 0; j < 8; ++j) {
        int d = 32 * kc + 2 * j + 16 * hf;   // A-layout K index (even)
        float2 f = *(const float2*)(qrow + d);
        qa[kc].u32[j] = pkbf(f.x * qscale, f.y * qscale);
      }
  }

  // ---- base mask bits, packed into 2 dwords: bit (c&3)*8+p of (c<4 ? lo : hi) ----
  unsigned mmlo = 0, mmhi = 0;
#pragma unroll
  for (int c = 0; c < 8; ++c) {
#pragma unroll
    for (int p = 0; p < 8; ++p) {
      int tr = w * 16 + p + 8 * hf;     // row t within sweep
      int tc = c * 16 + n;              // col t' within sweep
      unsigned bit = maskp[tr * NT + tc] ? 1u : 0u;
      if (c < 4) mmlo |= bit << ((c & 3) * 8 + p);
      else       mmhi |= bit << ((c & 3) * 8 + p);
    }
  }

  const int myTag = c1[b * NS + s];

  // ---- flash-attention state (m_/l_ live in the base-2 domain) ----
  v8f acc[8];
#pragma unroll
  for (int c = 0; c < 8; ++c) acc[c] = vzero8();
  float m_[8], l_[8];
#pragma unroll
  for (int p = 0; p < 8; ++p) { m_[p] = -1e30f; l_[p] = 0.0f; }

  for (int sp = 0; sp < NS; ++sp) {
    if (c1[b * NS + sp] != myTag) continue;   // block-uniform: whole sweep masked out

    __syncthreads();                          // previous V reads done
    // ---- stage K block [t'][d] as packed bf16 dwords ----
    {
      const float* kbase = k + ((size_t)(b * NS + sp)) * NT * ND;
#pragma unroll
      for (int it = 0; it < 16; ++it) {
        int i = tid + it * 256;
        int tp = i >> 5, d4 = (i & 31) * 4;
        float4 f = *(const float4*)(kbase + tp * ND + d4);
        unsigned* dst = &sKV[tp * (ND / 2) + (d4 >> 1)];
        dst[0] = pkbf(f.x, f.y);
        dst[1] = pkbf(f.z, f.w);
      }
      // prefetch next sweep's K block into cache (global_prefetch_b8)
      if (sp + 1 < NS) {
        const float* knext = k + ((size_t)(b * NS + sp + 1)) * NT * ND;
        __builtin_prefetch(knext + tid * 32, 0, 1);
        __builtin_prefetch(knext + (tid + 256) * 32, 0, 1);
      }
    }
    __syncthreads();

    // ---- scores: kc outer / c inner so all 8 B-tile loads per kc batch up ----
    v8f st[8];
#pragma unroll
    for (int kc = 0; kc < 4; ++kc) {
#pragma unroll
      for (int c = 0; c < 8; ++c) {
        BFU bu;  // B operand = K^T chunk: K index = d, N = t'  (pre-packed dwords)
        const unsigned* src = &sKV[(c * 16 + n) * (ND / 2) + 16 * kc + 8 * hf];
        uint4 r0 = *(const uint4*)(src);
        uint4 r1 = *(const uint4*)(src + 4);
        bu.u32[0] = r0.x; bu.u32[1] = r0.y; bu.u32[2] = r0.z; bu.u32[3] = r0.w;
        bu.u32[4] = r1.x; bu.u32[5] = r1.y; bu.u32[6] = r1.z; bu.u32[7] = r1.w;
        st[c] = __builtin_amdgcn_wmma_f32_16x16x32_bf16(
            false, qa[kc].v, false, bu.v, (short)0,
            (kc == 0) ? vzero8() : st[c], false, false);
      }
    }
    // mask (scores already scaled via Q); constant shifts after unroll
#pragma unroll
    for (int c = 0; c < 8; ++c) {
      unsigned mreg = (c < 4) ? mmlo : mmhi;
#pragma unroll
      for (int p = 0; p < 8; ++p)
        if (!((mreg >> ((c & 3) * 8 + p)) & 1u)) st[c][p] = -1e9f;
    }
    __syncthreads();                          // all waves done reading K

    // ---- stage V block TRANSPOSED [d][t'] as packed bf16 dwords,
    //      honoring the time-major scramble: v_row(t') = value[b, t'%16, 8*sp + t'/16]
    {
#pragma unroll
      for (int it = 0; it < 8; ++it) {
        int i = tid + it * 256;
        int ph = i >> 5;                 // t' pair index 0..63
        int d4 = (i & 31) * 4;
        int t0 = 2 * ph, t1 = 2 * ph + 1;
        const float* vr0 =
            v + (((size_t)(b * NS + (t0 & 15))) * NT + (8 * sp + (t0 >> 4))) * ND;
        const float* vr1 =
            v + (((size_t)(b * NS + (t1 & 15))) * NT + (8 * sp + (t1 >> 4))) * ND;
        float4 f0 = *(const float4*)(vr0 + d4);
        float4 f1 = *(const float4*)(vr1 + d4);
        sKV[(d4 + 0) * (NT / 2) + ph] = pkbf(f0.x, f1.x);
        sKV[(d4 + 1) * (NT / 2) + ph] = pkbf(f0.y, f1.y);
        sKV[(d4 + 2) * (NT / 2) + ph] = pkbf(f0.z, f1.z);
        sKV[(d4 + 3) * (NT / 2) + ph] = pkbf(f0.w, f1.w);
      }
    }
    __syncthreads();

    // ---- online softmax (base-2 domain: one v_exp_f32 per element) ----
#pragma unroll
    for (int p = 0; p < 8; ++p) {
      float x = -1e30f;
#pragma unroll
      for (int c = 0; c < 8; ++c) x = fmaxf(x, st[c][p]);
      x = redmax16(x);
      float mn = fmaxf(m_[p], x);
      float al = __builtin_amdgcn_exp2f(m_[p] - mn);
      m_[p] = mn;
      float rs = 0.0f;
#pragma unroll
      for (int c = 0; c < 8; ++c) {
        float e = __builtin_amdgcn_exp2f(st[c][p] - mn);
        st[c][p] = e;
        rs += e;
      }
      rs = redsum16(rs);
      l_[p] = l_[p] * al + rs;
#pragma unroll
      for (int c = 0; c < 8; ++c) acc[c][p] *= al;
    }

    // ---- write P (C-layout) to wave-private LDS as f32 ----
#pragma unroll
    for (int c = 0; c < 8; ++c)
#pragma unroll
      for (int p = 0; p < 8; ++p)
        sP[w][(p + 8 * hf) * NT + c * 16 + n] = st[c][p];

    // ---- P @ V : re-read P as A operand (pack on read), V as B operand ----
#pragma unroll
    for (int kc = 0; kc < 4; ++kc) {
      BFU pa;
      const float* pw = &sP[w][n * NT + 32 * kc + 8 * hf];
      float4 a0 = *(const float4*)(pw);
      float4 a1 = *(const float4*)(pw + 4);
      float4 a2 = *(const float4*)(pw + 16);
      float4 a3 = *(const float4*)(pw + 20);
      pa.u32[0] = pkbf(a0.x, a0.y); pa.u32[1] = pkbf(a0.z, a0.w);
      pa.u32[2] = pkbf(a1.x, a1.y); pa.u32[3] = pkbf(a1.z, a1.w);
      pa.u32[4] = pkbf(a2.x, a2.y); pa.u32[5] = pkbf(a2.z, a2.w);
      pa.u32[6] = pkbf(a3.x, a3.y); pa.u32[7] = pkbf(a3.z, a3.w);
#pragma unroll
      for (int c = 0; c < 8; ++c) {
        BFU vb;   // B operand: K = t', N = d (pre-packed dwords, table k order, as P's A operand)
        const unsigned* src = &sKV[(c * 16 + n) * (NT / 2) + 16 * kc + 4 * hf];
        uint4 r0 = *(const uint4*)(src);
        uint4 r1 = *(const uint4*)(src + 8);
        vb.u32[0] = r0.x; vb.u32[1] = r0.y; vb.u32[2] = r0.z; vb.u32[3] = r0.w;
        vb.u32[4] = r1.x; vb.u32[5] = r1.y; vb.u32[6] = r1.z; vb.u32[7] = r1.w;
        acc[c] = __builtin_amdgcn_wmma_f32_16x16x32_bf16(false, pa.v, false, vb.v,
                                                         (short)0, acc[c], false, false);
      }
    }
  }

  // ---- classifier weights loaded HERE (not held live across the main loop) ----
  float wv[8][3], bcv[3];
#pragma unroll
  for (int c = 0; c < 8; ++c)
#pragma unroll
    for (int cl = 0; cl < 3; ++cl) wv[c][cl] = Wc[(c * 16 + n) * 3 + cl];
#pragma unroll
  for (int cl = 0; cl < 3; ++cl) bcv[cl] = bc[cl];

  // ---- classifier + log10 Bayes factors + deadzone, per row ----
  float s3[3] = {0.0f, 0.0f, 0.0f};
#pragma unroll
  for (int p = 0; p < 8; ++p) {
    float linv = __builtin_amdgcn_rcpf(l_[p]);
    float lg[3];
#pragma unroll
    for (int cl = 0; cl < 3; ++cl) {
      float t = 0.0f;
#pragma unroll
      for (int c = 0; c < 8; ++c) t += acc[c][p] * wv[c][cl];
      lg[cl] = redsum16(t * linv) + bcv[cl];
    }
    float mx = fmaxf(lg[0], fmaxf(lg[1], lg[2]));
    float e0 = __expf(lg[0] - mx), e1 = __expf(lg[1] - mx), e2 = __expf(lg[2] - mx);
    float inv = __builtin_amdgcn_rcpf(e0 + e1 + e2);
    const float L10 = 0.43429448190325176f;
    float g0 = __logf(e0 * inv + 1e-8f) * L10;
    float g1 = __logf(e1 * inv + 1e-8f) * L10;
    float g2 = __logf(e2 * inv + 1e-8f) * L10;
    float tc_ = g1 - g0;
    float bc_ = g2 - g0;
    float bt_ = bc_ - tc_;                     // before clamping, as in reference
    tc_ = fminf(1.0f, fmaxf(-1.0f, tc_));
    bc_ = fminf(1.0f, fmaxf(-1.0f, bc_));
    bt_ = fminf(1.0f, fmaxf(-1.0f, bt_));
    s3[0] += (fabsf(tc_) > 0.1f) ? tc_ : 0.0f;  // deadzone
    s3[1] += (fabsf(bc_) > 0.1f) ? bc_ : 0.0f;
    s3[2] += (fabsf(bt_) > 0.1f) ? bt_ : 0.0f;
  }
  // lanes 0 and 16 hold the per-half sums (rows p / p+8); combine, lane0 commits
#pragma unroll
  for (int cl = 0; cl < 3; ++cl) {
    float val = s3[cl];
    val += __shfl_xor(val, 16, 32);
    if (lane == 0) atomicAdd(&sums[b * 3 + cl], val);
  }
}

__global__ void fetal_finalize_kernel(const float* __restrict__ sums,
                                      float* __restrict__ out) {
  int i = threadIdx.x;
  if (i < NB * 3) {
    float x = fminf(2.0f, fmaxf(-2.0f, sums[i]));
    float odds = __expf(x * 2.302585092994046f);  // 10^x
    out[i] = odds / (1.0f + odds);
  }
}

extern "C" void kernel_launch(void* const* d_in, const int* in_sizes, int n_in,
                              void* d_out, int out_size, void* d_ws, size_t ws_size,
                              hipStream_t stream) {
  const float*         q     = (const float*)d_in[0];
  const float*         k     = (const float*)d_in[1];
  const float*         v     = (const float*)d_in[2];
  const float*         Wc    = (const float*)d_in[3];
  const float*         bc    = (const float*)d_in[4];
  const unsigned char* maskp = (const unsigned char*)d_in[5];
  const int*           c1    = (const int*)d_in[6];
  float* sums = (float*)d_ws;
  float* out  = (float*)d_out;

  fetal_zero_kernel<<<1, 64, 0, stream>>>(sums);
  fetal_attn_kernel<<<dim3(NB * NS), dim3(256), 0, stream>>>(q, k, v, Wc, bc, maskp, c1, sums);
  fetal_finalize_kernel<<<1, 64, 0, stream>>>(sums, out);
}
